// GAT_1864015806544
// MI455X (gfx1250) — matrix operand order repaired and checked by reading, not compiled, not run on hardware
//
#include <hip/hip_runtime.h>
#include <stdint.h>

// ---------------- problem constants ----------------
#define N1c 262144
#define N2c 65536
#define N3c 16384
#define E1c 524288
#define E2c 262144
#define INDIMc 128
#define HDc 256      // HEADS*HID = 4*64
#define OUTDc 64

typedef __bf16 bf16;
typedef bf16  v16bf __attribute__((ext_vector_type(16)));
typedef float v8f   __attribute__((ext_vector_type(8)));
typedef unsigned int v4u  __attribute__((ext_vector_type(4)));
typedef int          v8i_ __attribute__((ext_vector_type(8)));
typedef int          v4i_ __attribute__((ext_vector_type(4)));

#if defined(__HIP_DEVICE_COMPILE__)
#if defined(__gfx1250__) && __has_builtin(__builtin_amdgcn_tensor_load_to_lds)
#define HAVE_TDM 1
#else
#define HAVE_TDM 0
#warning "CDNA5 TDM builtin NOT available on this toolchain; falling back to plain LDS staging"
#endif
#else
#define HAVE_TDM 0
#endif

// order-preserving float <-> uint encoding for atomicMax-based segment max
__device__ __forceinline__ unsigned ford(float f) {
    unsigned u = __float_as_uint(f);
    return (u & 0x80000000u) ? ~u : (u | 0x80000000u);
}
__device__ __forceinline__ float funord(unsigned o) {
    unsigned u = (o & 0x80000000u) ? (o ^ 0x80000000u) : ~o;
    return __uint_as_float(u);
}

// ---------------- prep: bf16 weights + collapsed attention matrices ----------------
__global__ void prep_kernel(const float* __restrict__ W1s, const float* __restrict__ W1d,
                            const float* __restrict__ att1s, const float* __restrict__ att1d,
                            const float* __restrict__ W2s, const float* __restrict__ W2d,
                            const float* __restrict__ att2s, const float* __restrict__ att2d,
                            bf16* __restrict__ W1s_bf, bf16* __restrict__ W2s_bf,
                            float* __restrict__ Wse, float* __restrict__ Wde,
                            float* __restrict__ w2se, float* __restrict__ w2de) {
    int tid = blockIdx.x * 256 + threadIdx.x;          // 32768 threads
    if (tid < INDIMc * HDc) W1s_bf[tid] = (bf16)W1s[tid];
    if (tid < HDc * OUTDc)  W2s_bf[tid] = (bf16)W2s[tid];
    if (tid < 512) {                                   // Wse[k][h] = sum_c W1s[k, h*64+c]*att1s[h,c]
        int k = tid >> 2, h2 = tid & 3;
        float acc = 0.f;
        for (int c = 0; c < 64; ++c) acc += W1s[k * HDc + h2 * 64 + c] * att1s[h2 * 64 + c];
        Wse[tid] = acc;
    } else if (tid < 1024) {
        int t = tid - 512; int k = t >> 2, h2 = t & 3;
        float acc = 0.f;
        for (int c = 0; c < 64; ++c) acc += W1d[k * HDc + h2 * 64 + c] * att1d[h2 * 64 + c];
        Wde[t] = acc;
    } else if (tid < 1280) {
        int k = tid - 1024;
        float acc = 0.f;
        for (int c = 0; c < 64; ++c) acc += W2s[k * 64 + c] * att2s[c];
        w2se[k] = acc;
    } else if (tid < 1536) {
        int k = tid - 1280;
        float acc = 0.f;
        for (int c = 0; c < 64; ++c) acc += W2d[k * 64 + c] * att2d[c];
        w2de[k] = acc;
    }
}

// ---------------- tiny init kernels ----------------
__global__ void fill_u32(unsigned* p, unsigned v, long n) {
    long i = (long)blockIdx.x * 256 + threadIdx.x;
    if (i < n) p[i] = v;
}
__global__ void fill_f32(float* p, float v, long n) {
    long i = (long)blockIdx.x * 256 + threadIdx.x;
    if (i < n) p[i] = v;
}
__global__ void init_bias(float* __restrict__ dst, const float* __restrict__ bias, long n, int C) {
    long i = (long)blockIdx.x * 256 + threadIdx.x;
    if (i < n) dst[i] = bias[i % C];
}

// ---------------- WMMA bf16 GEMM: C[M,N] = A[M,K] @ B[K,N] ----------------
// A: f32 row-major (converted on the fly), B: bf16 row-major in global.
// B panel is DMA'd into LDS by the Tensor Data Mover (count=1 2D descriptor),
// then transposed in-place through registers to the WMMA-B-friendly [N][K]
// layout. 256 threads = 8 waves; block computes 128 x N.
template <int K, int N>
__global__ __launch_bounds__(256) void gemm_wmma_bf16(const float* __restrict__ A,
                                                      const bf16* __restrict__ Bg,
                                                      float* __restrict__ C) {
    __shared__ bf16 Bt[N * K];  // final layout: Bt[n*K + k]
    const int tid = threadIdx.x;

#if HAVE_TDM
    // ---- TDM: DMA B (row-major [K][N] bf16) into LDS ----
    if (tid == 0) {
        unsigned lds = (unsigned)(size_t)(void*)&Bt[0];          // LDS byte offset
        unsigned long long ga = (unsigned long long)(size_t)Bg;  // global byte address
        v4u g0;
        g0[0] = 1u;                                   // count=1, user mode, no gather
        g0[1] = lds;                                  // lds_addr
        g0[2] = (unsigned)(ga & 0xffffffffu);         // global_addr[31:0]
        g0[3] = (unsigned)((ga >> 32) & 0x01ffffffu)  // global_addr[56:32]
                | 0x80000000u;                        // type = 2 ("image")
        v8i_ g1;
        g1[0] = 1 << 16;                              // wg_mask=0, data_size=1 (2 bytes)
        g1[1] = (N & 0xffff) << 16;                   // tensor_dim0[15:0]
        g1[2] = (N >> 16) | ((K & 0xffff) << 16);     // tensor_dim0[31:16] | tensor_dim1[15:0]
        g1[3] = (K >> 16) | ((N & 0xffff) << 16);     // tensor_dim1[31:16] | tile_dim0
        g1[4] = (K & 0xffff);                         // tile_dim1 | tile_dim2=0
        g1[5] = N;                                    // tensor_dim0_stride[31:0]
        g1[6] = 0;                                    // stride0[47:32] | tensor_dim1_stride[15:0]
        g1[7] = 0;                                    // tensor_dim1_stride[47:16]
        v4i_ g2 = {};                                 // 2D tensor: higher dims zero
        v4i_ g3 = {};
#if defined(__clang_major__) && (__clang_major__ >= 23)
        v8i_ g4 = {};
        __builtin_amdgcn_tensor_load_to_lds(g0, g1, g2, g3, g4, 0);
#else
        __builtin_amdgcn_tensor_load_to_lds(g0, g1, g2, g3, 0);
#endif
        __builtin_amdgcn_s_wait_tensorcnt(0);
    }
    __syncthreads();
    // ---- in-place transpose [K][N] -> [N][K] through registers ----
    constexpr int PT = (K * N) / 512;   // uint32 (bf16 pair) per thread
    uint32_t r[PT];
    const uint32_t* Braw32 = (const uint32_t*)Bt;
#pragma unroll
    for (int i = 0; i < PT; ++i) r[i] = Braw32[tid * PT + i];
    __syncthreads();
#pragma unroll
    for (int i = 0; i < PT; ++i) {
        int e = (tid * PT + i) * 2;     // linear row-major element index
        int k = e / N, n = e % N;       // pair is (k,n), (k,n+1)
        Bt[n * K + k]       = __builtin_bit_cast(bf16, (unsigned short)(r[i] & 0xffffu));
        Bt[(n + 1) * K + k] = __builtin_bit_cast(bf16, (unsigned short)(r[i] >> 16));
    }
#else
    // ---- fallback: cooperative global load + transpose ----
    const uint32_t* Bg32 = (const uint32_t*)Bg;
    for (int idx = tid; idx < (K * N) / 2; idx += 256) {
        uint32_t w = Bg32[idx];
        int k = (idx * 2) / N;
        int n = (idx * 2) % N;
        Bt[n * K + k]       = __builtin_bit_cast(bf16, (unsigned short)(w & 0xffffu));
        Bt[(n + 1) * K + k] = __builtin_bit_cast(bf16, (unsigned short)(w >> 16));
    }
#endif
    __syncthreads();

    const int wave = tid >> 5;
    const int lane = tid & 31;
    const int lh   = lane >> 4;   // lane half (K-offset selector)
    const int lm   = lane & 15;   // M row (A) / N col (B)
    const long m   = (long)blockIdx.x * 128 + wave * 16 + lm;

    // preload all A fragments across K (16-bit A layout: V0-3 K=0..7(+8*lh), V4-7 K=16..23(+8*lh))
    constexpr int KT = K / 32;
    v16bf afrag[KT];
    const float* Arow = A + m * K;
    __builtin_prefetch(Arow, 0, 1);     // global_prefetch_b8
#pragma unroll
    for (int kt = 0; kt < KT; ++kt) {
        v16bf a;
#pragma unroll
        for (int v = 0; v < 8; ++v) {
            int kk = kt * 32 + 8 * lh + (v < 4 ? 2 * v : 16 + 2 * (v - 4));
            float2 f = *(const float2*)(Arow + kk);
            a[2 * v]     = (bf16)f.x;
            a[2 * v + 1] = (bf16)f.y;
        }
        afrag[kt] = a;
    }

    for (int nt = 0; nt < N / 16; ++nt) {
        v8f acc = {};
        const int n = nt * 16 + lm;
#pragma unroll
        for (int kt = 0; kt < KT; ++kt) {
            // 16-bit B layout: lane=n, V0-3 K=0..7(+8*lh), V4-7 K=16..23(+8*lh) (pairs contiguous in Bt)
            union { v16bf v; uint32_t u[8]; } bu;
            const uint32_t* brow32 = (const uint32_t*)&Bt[n * K + kt * 32 + 8 * lh];
#pragma unroll
            for (int v = 0; v < 8; ++v) bu.u[v] = brow32[v < 4 ? v : v + 4];
            acc = __builtin_amdgcn_wmma_f32_16x16x32_bf16(false, afrag[kt], false, bu.v,
                                                          (short)0, acc, false, false);
        }
        // C/D layout: VGPR r holds row (8*lh + r), col lm
        const long mbase = (long)blockIdx.x * 128 + wave * 16 + 8 * lh;
#pragma unroll
        for (int r = 0; r < 8; ++r) {
            C[(mbase + r) * N + nt * 16 + lm] = acc[r];
        }
    }
}

// ---------------- attention logits, layer 1: a = x @ Weff  (wave per row) ----------------
__global__ __launch_bounds__(256) void att1_coeff(const float* __restrict__ x,
                                                  const float* __restrict__ Wse,
                                                  const float* __restrict__ Wde,
                                                  float* __restrict__ a1s, float* __restrict__ a1d) {
    int wave = threadIdx.x >> 5, lane = threadIdx.x & 31;
    long row = (long)blockIdx.x * 8 + wave;
    const float* xr = x + row * INDIMc;
    float s[4] = {0.f, 0.f, 0.f, 0.f};
    float d[4] = {0.f, 0.f, 0.f, 0.f};
#pragma unroll
    for (int j = 0; j < 4; ++j) {
        int k = j * 32 + lane;
        float v = xr[k];
        const float* we = Wse + k * 4;
        const float* wd = Wde + k * 4;
#pragma unroll
        for (int h2 = 0; h2 < 4; ++h2) { s[h2] += v * we[h2]; d[h2] += v * wd[h2]; }
    }
#pragma unroll
    for (int h2 = 0; h2 < 4; ++h2)
        for (int off = 16; off; off >>= 1) {
            s[h2] += __shfl_xor(s[h2], off, 32);
            d[h2] += __shfl_xor(d[h2], off, 32);
        }
    if (lane == 0) {
#pragma unroll
        for (int h2 = 0; h2 < 4; ++h2) a1s[row * 4 + h2] = s[h2];
        if (row < N2c)
#pragma unroll
            for (int h2 = 0; h2 < 4; ++h2) a1d[row * 4 + h2] = d[h2];
    }
}

// ---------------- attention logits, layer 2 ----------------
__global__ __launch_bounds__(256) void att2_coeff(const float* __restrict__ hn,
                                                  const float* __restrict__ w2se,
                                                  const float* __restrict__ w2de,
                                                  float* __restrict__ a2s, float* __restrict__ a2d) {
    int wave = threadIdx.x >> 5, lane = threadIdx.x & 31;
    long row = (long)blockIdx.x * 8 + wave;
    const float* hr = hn + row * HDc;
    float ss = 0.f, sd = 0.f;
#pragma unroll
    for (int j = 0; j < 8; ++j) {
        int c = j * 32 + lane;
        float v = hr[c];
        ss += v * w2se[c];
        sd += v * w2de[c];
    }
    for (int off = 16; off; off >>= 1) {
        ss += __shfl_xor(ss, off, 32);
        sd += __shfl_xor(sd, off, 32);
    }
    if (lane == 0) {
        a2s[row] = ss;
        if (row < N3c) a2d[row] = sd;
    }
}

// ---------------- edge pass 1: leaky-relu logits + segment max ----------------
__global__ void edge_logits_max(const int* __restrict__ src, const int* __restrict__ dst,
                                const float* __restrict__ as, const float* __restrict__ ad,
                                float* __restrict__ ebuf, unsigned* __restrict__ emax,
                                int E, int H) {
    int e = blockIdx.x * 256 + threadIdx.x;
    if (e >= E) return;
    int s = src[e], d = dst[e];
    for (int h2 = 0; h2 < H; ++h2) {
        float v = as[(long)s * H + h2] + ad[(long)d * H + h2];
        v = v > 0.f ? v : 0.2f * v;
        ebuf[(long)e * H + h2] = v;
        atomicMax(&emax[(long)d * H + h2], ford(v));
    }
}

// ---------------- edge pass 2: exp + segment sum ----------------
__global__ void edge_softmax_sum(const int* __restrict__ dst,
                                 float* __restrict__ ebuf, const unsigned* __restrict__ emax,
                                 float* __restrict__ denom, int E, int H) {
    int e = blockIdx.x * 256 + threadIdx.x;
    if (e >= E) return;
    int d = dst[e];
    for (int h2 = 0; h2 < H; ++h2) {
        float p = __expf(ebuf[(long)e * H + h2] - funord(emax[(long)d * H + h2]));
        ebuf[(long)e * H + h2] = p;
        atomicAdd(&denom[(long)d * H + h2], p);
    }
}

// ---------------- edge pass 3: normalized message scatter (wave per edge) ----------------
__global__ __launch_bounds__(256) void edge_scatter1(const int* __restrict__ src, const int* __restrict__ dst,
                                                     const float* __restrict__ p, const float* __restrict__ denom,
                                                     const float* __restrict__ hsrc, float* __restrict__ hout) {
    int wave = threadIdx.x >> 5, lane = threadIdx.x & 31;
    long e = (long)blockIdx.x * 8 + wave;
    int s = src[e], d = dst[e];
    float alpha[4];
#pragma unroll
    for (int h2 = 0; h2 < 4; ++h2)
        alpha[h2] = p[e * 4 + h2] / fmaxf(denom[(long)d * 4 + h2], 1e-16f);
    const float* hs = hsrc + (long)s * HDc;
    float* ho = hout + (long)d * HDc;
#pragma unroll
    for (int j = 0; j < 8; ++j) {
        int c = j * 32 + lane;                 // head = c/64 = j>>1
        atomicAdd(&ho[c], hs[c] * alpha[j >> 1]);
    }
}

__global__ __launch_bounds__(256) void edge_scatter2(const int* __restrict__ src, const int* __restrict__ dst,
                                                     const float* __restrict__ p, const float* __restrict__ denom,
                                                     const float* __restrict__ hsrc, float* __restrict__ out) {
    int wave = threadIdx.x >> 5, lane = threadIdx.x & 31;
    long e = (long)blockIdx.x * 8 + wave;
    int s = src[e], d = dst[e];
    float alpha = p[e] / fmaxf(denom[d], 1e-16f);
    const float* hs = hsrc + (long)s * OUTDc;
    float* ho = out + (long)d * OUTDc;
#pragma unroll
    for (int j = 0; j < 2; ++j) {
        int c = j * 32 + lane;
        atomicAdd(&ho[c], hs[c] * alpha);
    }
}

// ---------------- BatchNorm (eval) + ELU, in place ----------------
__global__ void bn_elu(float* __restrict__ hbuf, const float* __restrict__ gamma,
                       const float* __restrict__ beta, const float* __restrict__ mean,
                       const float* __restrict__ var) {
    long tid = (long)blockIdx.x * 256 + threadIdx.x;
    int c = (int)(tid & (HDc - 1));
    float v = hbuf[tid];
    v = (v - mean[c]) * gamma[c] * rsqrtf(var[c] + 1e-5f) + beta[c];
    hbuf[tid] = v > 0.f ? v : (__expf(v) - 1.f);
}

// ---------------- host launcher ----------------
extern "C" void kernel_launch(void* const* d_in, const int* in_sizes, int n_in,
                              void* d_out, int out_size, void* d_ws, size_t ws_size,
                              hipStream_t stream) {
    (void)in_sizes; (void)n_in; (void)out_size; (void)ws_size;
    const float* x      = (const float*)d_in[0];
    const float* W1s    = (const float*)d_in[1];
    const float* W1d    = (const float*)d_in[2];
    const float* att1s  = (const float*)d_in[3];
    const float* att1d  = (const float*)d_in[4];
    const float* b1     = (const float*)d_in[5];
    const float* gamma  = (const float*)d_in[6];
    const float* beta   = (const float*)d_in[7];
    const float* rmean  = (const float*)d_in[8];
    const float* rvar   = (const float*)d_in[9];
    const float* W2s    = (const float*)d_in[10];
    const float* W2d    = (const float*)d_in[11];
    const float* att2s  = (const float*)d_in[12];
    const float* att2d  = (const float*)d_in[13];
    const float* b2     = (const float*)d_in[14];
    const int*   src1   = (const int*)d_in[15];
    const int*   dst1   = (const int*)d_in[16];
    const int*   src2   = (const int*)d_in[17];
    const int*   dst2   = (const int*)d_in[18];
    float* out = (float*)d_out;

    // workspace carve-up
    char* ws = (char*)d_ws;
    size_t off = 0;
    auto alloc = [&](size_t bytes) { char* p = ws + off; off = (off + bytes + 255) & ~(size_t)255; return p; };
    float*    hsrc1  = (float*)alloc((size_t)N1c * HDc * 4);   // 256 MB
    float*    hbuf   = (float*)alloc((size_t)N2c * HDc * 4);   // 64 MB (layer-1 output)
    float*    hsrc2  = (float*)alloc((size_t)N2c * OUTDc * 4); // 16 MB
    float*    a1s    = (float*)alloc((size_t)N1c * 4 * 4);
    float*    a1d    = (float*)alloc((size_t)N2c * 4 * 4);
    unsigned* emax1  = (unsigned*)alloc((size_t)N2c * 4 * 4);
    float*    den1   = (float*)alloc((size_t)N2c * 4 * 4);
    float*    ebuf1  = (float*)alloc((size_t)E1c * 4 * 4);
    float*    a2s    = (float*)alloc((size_t)N2c * 4);
    float*    a2d    = (float*)alloc((size_t)N3c * 4);
    unsigned* emax2  = (unsigned*)alloc((size_t)N3c * 4);
    float*    den2   = (float*)alloc((size_t)N3c * 4);
    float*    ebuf2  = (float*)alloc((size_t)E2c * 4);
    bf16*     W1sbf  = (bf16*)alloc((size_t)INDIMc * HDc * 2);
    bf16*     W2sbf  = (bf16*)alloc((size_t)HDc * OUTDc * 2);
    float*    Wse    = (float*)alloc(512 * 4);
    float*    Wde    = (float*)alloc(512 * 4);
    float*    w2se   = (float*)alloc(256 * 4);
    float*    w2de   = (float*)alloc(256 * 4);

    const unsigned NEG_INF_ORD = 0x007FFFFFu;  // ford(-inf)

    // prep + init
    prep_kernel<<<128, 256, 0, stream>>>(W1s, W1d, att1s, att1d, W2s, W2d, att2s, att2d,
                                         W1sbf, W2sbf, Wse, Wde, w2se, w2de);
    fill_u32<<<(N2c * 4) / 256, 256, 0, stream>>>(emax1, NEG_INF_ORD, (long)N2c * 4);
    fill_f32<<<(N2c * 4) / 256, 256, 0, stream>>>(den1, 0.f, (long)N2c * 4);
    fill_u32<<<N3c / 256, 256, 0, stream>>>(emax2, NEG_INF_ORD, (long)N3c);
    fill_f32<<<N3c / 256, 256, 0, stream>>>(den2, 0.f, (long)N3c);
    init_bias<<<((long)N2c * HDc) / 256, 256, 0, stream>>>(hbuf, b1, (long)N2c * HDc, HDc);
    init_bias<<<((long)N3c * OUTDc) / 256, 256, 0, stream>>>(out, b2, (long)N3c * OUTDc, OUTDc);

    // layer 1
    gemm_wmma_bf16<INDIMc, HDc><<<N1c / 128, 256, 0, stream>>>(x, W1sbf, hsrc1);
    att1_coeff<<<N1c / 8, 256, 0, stream>>>(x, Wse, Wde, a1s, a1d);
    edge_logits_max<<<E1c / 256, 256, 0, stream>>>(src1, dst1, a1s, a1d, ebuf1, emax1, E1c, 4);
    edge_softmax_sum<<<E1c / 256, 256, 0, stream>>>(dst1, ebuf1, emax1, den1, E1c, 4);
    edge_scatter1<<<E1c / 8, 256, 0, stream>>>(src1, dst1, ebuf1, den1, hsrc1, hbuf);
    bn_elu<<<((long)N2c * HDc) / 256, 256, 0, stream>>>(hbuf, gamma, beta, rmean, rvar);

    // layer 2
    gemm_wmma_bf16<HDc, OUTDc><<<N2c / 128, 256, 0, stream>>>(hbuf, W2sbf, hsrc2);
    att2_coeff<<<N2c / 8, 256, 0, stream>>>(hbuf, w2se, w2de, a2s, a2d);
    edge_logits_max<<<E2c / 256, 256, 0, stream>>>(src2, dst2, a2s, a2d, ebuf2, emax2, E2c, 1);
    edge_softmax_sum<<<E2c / 256, 256, 0, stream>>>(dst2, ebuf2, emax2, den2, E2c, 1);
    edge_scatter2<<<E2c / 8, 256, 0, stream>>>(src2, dst2, ebuf2, den2, hsrc2, out);
}
